// MyLSTM_8143257994011
// MI455X (gfx1250) — matrix operand order repaired and checked by reading, not compiled, not run on hardware
//
#include <hip/hip_runtime.h>

// ---------------------------------------------------------------------------
// 2-layer LSTM, T=512, B=64, I=512, H=1024 — persistent WMMA kernel for
// MI455X (gfx1250, wave32).
//   * bf16 activations + weights (fp32 accumulate), weights resident in L2
//   * per-WG A-panel staged into LDS via global_load_async_to_lds_b128
//   * each wave: 2 M-tiles x 1 N-tile, B operand shared across 2 WMMAs/k-step
//   * grid-wide sense-reversing barrier between the 4 phases of each timestep
// ---------------------------------------------------------------------------

constexpr int T_STEPS = 512;
constexpr int BATCH   = 64;
constexpr int IN_DIM  = 512;
constexpr int HID     = 1024;
constexpr int GATES   = 4 * HID;               // 4096
constexpr int NBLK    = 64;                    // persistent workgroups
constexpr int NTHR    = 256;                   // 8 wave32 waves / WG
constexpr int LDS_BYTES = 2 * BATCH * (HID + HID); // 256 KB (layer-1 panel)

typedef __bf16 bf16;
typedef __attribute__((ext_vector_type(16))) __bf16 v16bf;
typedef __attribute__((ext_vector_type(8)))  __bf16 v8bf;
typedef __attribute__((ext_vector_type(8)))  float  v8f;

__device__ __forceinline__ float fsigmoid(float x) {
  return 1.0f / (1.0f + __expf(-x));
}
__device__ __forceinline__ float ftanh(float x) {
  float e = __expf(-2.0f * x);
  return (1.0f - e) / (1.0f + e);
}

// ---------------------------------------------------------------------------
// Async-stage a contiguous bf16 slab (global -> LDS), 16B granules.
// Trip count is uniform across the block (nelem/8 is a multiple of NTHR here),
// so EXEC stays full and issue is fully coalesced.
// ---------------------------------------------------------------------------
__device__ __forceinline__ void stage_async(const bf16* __restrict__ g,
                                            bf16* __restrict__ sm, int nelem) {
  const int n16 = nelem >> 3;                      // 16-byte chunks
  for (int i = threadIdx.x; i < n16; i += NTHR) {
    unsigned    lds = (unsigned)(size_t)(sm + (size_t)i * 8);  // addr[31:0] = LDS offset
    const bf16* src = g + (size_t)i * 8;
    asm volatile("global_load_async_to_lds_b128 %0, %1, off"
                 :: "v"(lds), "v"(src) : "memory");
  }
}

// ---------------------------------------------------------------------------
// K-segment MMA: A panel in LDS (row-major [64][K] bf16), weights Bw row-major
// [N][K] bf16 in global (L2-resident).  One N-tile, two M-tiles per wave; the
// B fragment is reused by both WMMAs of each k-step.
//
// CDNA5 VGPR striping (16x16x32 bf16):
//  A: lane l holds row M=l&15, K-chunks {0..7,16..23} (l<16) / {8..15,24..31}
//  B: lane l holds col N=l&15, K-chunks {0..7,16..23} (l<16) / {8..15,24..31}
// ---------------------------------------------------------------------------
__device__ __forceinline__ void mma_seg(const bf16* __restrict__ sm, int K,
                                        const bf16* __restrict__ Bw,
                                        int n0, int m0,
                                        v8f& acc0, v8f& acc1, int lane) {
  const int am  = lane & 15;
  const int ak0 = (lane >> 4) << 3;    // 0 or 8
  const int bn  = lane & 15;
  const int bk0 = (lane >> 4) << 3;    // 0 or 8
  const bf16* brow = Bw + (size_t)(n0 + bn) * K + bk0;
  const bf16* ar0  = sm + (size_t)(m0 + am) * K + ak0;
  const bf16* ar1  = ar0 + (size_t)16 * K;
  for (int k = 0; k < K; k += 32) {
    v8bf  blo  = *(const v8bf*)(brow + k);
    v8bf  bhi  = *(const v8bf*)(brow + k + 16);
    v16bf b    = __builtin_shufflevector(blo, bhi,
                 0, 1, 2, 3, 4, 5, 6, 7, 8, 9, 10, 11, 12, 13, 14, 15);
    v8bf  a0lo = *(const v8bf*)(ar0 + k);            // ds_load_b128
    v8bf  a0hi = *(const v8bf*)(ar0 + k + 16);
    v8bf  a1lo = *(const v8bf*)(ar1 + k);
    v8bf  a1hi = *(const v8bf*)(ar1 + k + 16);
    v16bf a0 = __builtin_shufflevector(a0lo, a0hi,
                 0, 1, 2, 3, 4, 5, 6, 7, 8, 9, 10, 11, 12, 13, 14, 15);
    v16bf a1 = __builtin_shufflevector(a1lo, a1hi,
                 0, 1, 2, 3, 4, 5, 6, 7, 8, 9, 10, 11, 12, 13, 14, 15);
    if (k + 32 < K) __builtin_prefetch(brow + k + 32, 0, 3);
    acc0 = __builtin_amdgcn_wmma_f32_16x16x32_bf16(false, a0, false, b,
                                                   (short)0, acc0, false, false);
    acc1 = __builtin_amdgcn_wmma_f32_16x16x32_bf16(false, a1, false, b,
                                                   (short)0, acc1, false, false);
  }
}

// gates[B x 4H] = A1 @ B1^T + A2 @ B2^T + bias  (this WG's 64-column slab)
__device__ __forceinline__ void gemm_phase_lds(
    const bf16* __restrict__ A1g, int K1, const bf16* __restrict__ A2g, int K2,
    const bf16* __restrict__ B1w, const bf16* __restrict__ B2w,
    const float* __restrict__ bias, float* __restrict__ gout,
    bf16* __restrict__ sm) {
  stage_async(A1g, sm, BATCH * K1);
  stage_async(A2g, sm + BATCH * K1, BATCH * K2);
  asm volatile("s_wait_asynccnt 0x0" ::: "memory");
  __syncthreads();

  const int wv   = threadIdx.x >> 5;
  const int lane = threadIdx.x & 31;
  const int n0   = blockIdx.x * 64 + (wv & 3) * 16;   // WG owns 64 gate columns
  const int m0   = (wv >> 2) * 32;                    // wave owns M tiles m0, m0+16
  v8f acc0 = {}, acc1 = {};
  mma_seg(sm,              K1, B1w, n0, m0, acc0, acc1, lane);
  mma_seg(sm + BATCH * K1, K2, B2w, n0, m0, acc0, acc1, lane);

  // C/D layout: VGPR r -> (M = m0 + r + (lane<16?0:8), N = n0 + (lane&15))
  const int   n  = n0 + (lane & 15);
  const float bv = bias[n];
  const int   mo = (lane >> 4) << 3;
#pragma unroll
  for (int r = 0; r < 8; ++r) {
    gout[(size_t)(m0 + mo + r) * GATES + n]      = acc0[r] + bv;
    gout[(size_t)(m0 + 16 + mo + r) * GATES + n] = acc1[r] + bv;
  }
  __syncthreads();   // smem reused by next phase
}

// Elementwise LSTM cell over [B x H]; gate order (i, f, g, o) per reference.
// h is kept in bf16 (it is only ever consumed as a WMMA A-operand).
__device__ __forceinline__ void cell_phase(const float* __restrict__ g,
                                           float* __restrict__ c,
                                           bf16* __restrict__ h,
                                           float* __restrict__ out,
                                           int tidg) {
  for (int idx = tidg; idx < BATCH * HID; idx += NBLK * NTHR) {
    const int b = idx >> 10, j = idx & (HID - 1);
    const float* gr = g + (size_t)b * GATES;
    float ig = fsigmoid(gr[j]);
    float fg = fsigmoid(gr[HID + j]);
    float gg = ftanh(gr[2 * HID + j]);
    float og = fsigmoid(gr[3 * HID + j]);
    float cn = fg * c[idx] + ig * gg;
    float hn = og * ftanh(cn);
    c[idx] = cn;
    h[idx] = (bf16)hn;
    if (out) out[idx] = hn;
  }
}

// Sense-reversing generation barrier across all NBLK resident workgroups.
__device__ __forceinline__ void grid_barrier(unsigned* __restrict__ ctrl) {
  __syncthreads();
  if (threadIdx.x == 0) {
    __threadfence();
    unsigned gen  = __hip_atomic_load(&ctrl[1], __ATOMIC_ACQUIRE, __HIP_MEMORY_SCOPE_AGENT);
    unsigned prev = __hip_atomic_fetch_add(&ctrl[0], 1u, __ATOMIC_ACQ_REL, __HIP_MEMORY_SCOPE_AGENT);
    if (prev == (unsigned)(NBLK - 1)) {
      __hip_atomic_store(&ctrl[0], 0u, __ATOMIC_RELAXED, __HIP_MEMORY_SCOPE_AGENT);
      __hip_atomic_fetch_add(&ctrl[1], 1u, __ATOMIC_RELEASE, __HIP_MEMORY_SCOPE_AGENT);
    } else {
      while (__hip_atomic_load(&ctrl[1], __ATOMIC_ACQUIRE, __HIP_MEMORY_SCOPE_AGENT) == gen)
        __builtin_amdgcn_s_sleep(1);
    }
  }
  __syncthreads();
}

// ---------------------------------------------------------------------------
// Prep kernels
// ---------------------------------------------------------------------------
__global__ void k_f32_to_bf16(const float* __restrict__ s, bf16* __restrict__ d, int n) {
  int i = blockIdx.x * blockDim.x + threadIdx.x;
  if (i < n) d[i] = (bf16)s[i];
}
__global__ void k_bias_sum(const float* __restrict__ a, const float* __restrict__ b,
                           float* __restrict__ d, int n) {
  int i = blockIdx.x * blockDim.x + threadIdx.x;
  if (i < n) d[i] = a[i] + b[i];
}
__global__ void k_zero_f32(float* __restrict__ p, int n) {
  int i = blockIdx.x * blockDim.x + threadIdx.x;
  if (i < n) p[i] = 0.0f;
}
__global__ void k_zero_bf16(bf16* __restrict__ p, int n) {
  int i = blockIdx.x * blockDim.x + threadIdx.x;
  if (i < n) p[i] = (bf16)0.0f;
}
__global__ void k_init_ctrl(unsigned* __restrict__ ctrl) {
  if (threadIdx.x < 2) ctrl[threadIdx.x] = 0u;
}

// ---------------------------------------------------------------------------
// Persistent LSTM kernel
// ---------------------------------------------------------------------------
__global__ __launch_bounds__(NTHR) void k_lstm_persistent(
    const bf16* __restrict__ xb,
    const bf16* __restrict__ wih0, const bf16* __restrict__ whh0,
    const bf16* __restrict__ wih1, const bf16* __restrict__ whh1,
    const float* __restrict__ bias0, const float* __restrict__ bias1,
    bf16* __restrict__ h0, bf16* __restrict__ h1,
    float* __restrict__ c0, float* __restrict__ c1,
    float* __restrict__ g0, float* __restrict__ g1,
    float* __restrict__ out, unsigned* __restrict__ ctrl) {
  extern __shared__ bf16 smem[];
  const int tidg = blockIdx.x * NTHR + threadIdx.x;

  for (int t = 0; t < T_STEPS; ++t) {
    const bf16* xt = xb + (size_t)t * BATCH * IN_DIM;
    // layer 0: gates0 = x_t @ w_ih0^T + h0 @ w_hh0^T + bias0
    gemm_phase_lds(xt, IN_DIM, h0, HID, wih0, whh0, bias0, g0, smem);
    grid_barrier(ctrl);
    cell_phase(g0, c0, h0, nullptr, tidg);
    grid_barrier(ctrl);
    // layer 1: gates1 = h0 @ w_ih1^T + h1 @ w_hh1^T + bias1
    gemm_phase_lds(h0, HID, h1, HID, wih1, whh1, bias1, g1, smem);
    grid_barrier(ctrl);
    cell_phase(g1, c1, h1, out + (size_t)t * BATCH * HID, tidg);
    grid_barrier(ctrl);
  }

  // finals: out layout = [ out(T,B,H) | h_f(2,B,H) | c_f(2,B,H) ]
  float* hf = out + (size_t)T_STEPS * BATCH * HID;
  float* cf = hf + 2 * BATCH * HID;
  for (int i = tidg; i < BATCH * HID; i += NBLK * NTHR) {
    hf[i]               = (float)h0[i];
    hf[BATCH * HID + i] = (float)h1[i];
    cf[i]               = c0[i];
    cf[BATCH * HID + i] = c1[i];
  }
}

// ---------------------------------------------------------------------------
extern "C" void kernel_launch(void* const* d_in, const int* in_sizes, int n_in,
                              void* d_out, int out_size, void* d_ws, size_t ws_size,
                              hipStream_t stream) {
  (void)in_sizes; (void)n_in; (void)out_size; (void)ws_size;
  const float* x     = (const float*)d_in[0];
  const float* w_ih0 = (const float*)d_in[1];
  const float* w_hh0 = (const float*)d_in[2];
  const float* b_ih0 = (const float*)d_in[3];
  const float* b_hh0 = (const float*)d_in[4];
  const float* w_ih1 = (const float*)d_in[5];
  const float* w_hh1 = (const float*)d_in[6];
  const float* b_ih1 = (const float*)d_in[7];
  const float* b_hh1 = (const float*)d_in[8];
  float* out = (float*)d_out;

  char* p = (char*)d_ws;
  auto take = [&](size_t bytes) {
    char* r = p;
    p += (bytes + 255) & ~size_t(255);
    return r;
  };
  unsigned* ctrl = (unsigned*)take(256);
  bf16* wih0b = (bf16*)take((size_t)GATES * IN_DIM * sizeof(bf16));
  bf16* whh0b = (bf16*)take((size_t)GATES * HID * sizeof(bf16));
  bf16* wih1b = (bf16*)take((size_t)GATES * HID * sizeof(bf16));
  bf16* whh1b = (bf16*)take((size_t)GATES * HID * sizeof(bf16));
  bf16* xb    = (bf16*)take((size_t)T_STEPS * BATCH * IN_DIM * sizeof(bf16));
  float* bias0 = (float*)take((size_t)GATES * sizeof(float));
  float* bias1 = (float*)take((size_t)GATES * sizeof(float));
  bf16*  h0 = (bf16*)take((size_t)BATCH * HID * sizeof(bf16));
  bf16*  h1 = (bf16*)take((size_t)BATCH * HID * sizeof(bf16));
  float* c0 = (float*)take((size_t)BATCH * HID * sizeof(float));
  float* c1 = (float*)take((size_t)BATCH * HID * sizeof(float));
  float* g0 = (float*)take((size_t)BATCH * GATES * sizeof(float));
  float* g1 = (float*)take((size_t)BATCH * GATES * sizeof(float));

  auto cvt = [&](const float* s, bf16* d, int n) {
    k_f32_to_bf16<<<(n + 255) / 256, 256, 0, stream>>>(s, d, n);
  };
  cvt(w_ih0, wih0b, GATES * IN_DIM);
  cvt(w_hh0, whh0b, GATES * HID);
  cvt(w_ih1, wih1b, GATES * HID);
  cvt(w_hh1, whh1b, GATES * HID);
  cvt(x, xb, T_STEPS * BATCH * IN_DIM);
  k_bias_sum<<<(GATES + 255) / 256, 256, 0, stream>>>(b_ih0, b_hh0, bias0, GATES);
  k_bias_sum<<<(GATES + 255) / 256, 256, 0, stream>>>(b_ih1, b_hh1, bias1, GATES);
  k_zero_f32<<<(2 * BATCH * HID + 255) / 256, 256, 0, stream>>>(c0, 2 * BATCH * HID);
  k_zero_bf16<<<(2 * BATCH * HID + 255) / 256, 256, 0, stream>>>(h0, 2 * BATCH * HID);
  k_init_ctrl<<<1, 64, 0, stream>>>(ctrl);

  k_lstm_persistent<<<NBLK, NTHR, LDS_BYTES, stream>>>(
      xb, wih0b, whh0b, wih1b, whh1b, bias0, bias1,
      h0, h1, c0, c1, g0, g1, out, ctrl);
}
